// WaterNet0313_18537078849769
// MI455X (gfx1250) — matrix operand order repaired and checked by reading, not compiled, not run on hardware
//
#include <hip/hip_runtime.h>
#include <hip/hip_bf16.h>
#include <math.h>

#define NT   1000
#define NS   500
#define NGc  32
#define NHb  16
#define NRl  16
#define HSd  256
#define WARM 639

typedef __attribute__((ext_vector_type(16))) _Float16 v16h;
typedef __attribute__((ext_vector_type(8)))  float    v8f;

__device__ __forceinline__ float sigmoidf_(float x) { return 1.0f / (1.0f + expf(-x)); }

// ------------- weight prep: fp32 -> f16, pre-swizzled into WMMA B-fragment order -------------
// W1p: [16 ntiles][2 ksteps][32 lanes][16 e]  (K padded 38->64, N=256)
// W2p: [ 3 ntiles][8 ksteps][32 lanes][16 e]  (K=256, N=48)
// Fragment element e at lane L holds B[K][N] with K = ks*32 + (e/8)*16 + (L/16)*8 + e%8, N = ntile*16 + L%16.
__global__ void prep_weights(const float* __restrict__ Wt1, const float* __restrict__ Wt2,
                             _Float16* __restrict__ W1p, _Float16* __restrict__ W2p) {
  int i = blockIdx.x * blockDim.x + threadIdx.x;
  if (i < 16 * 2 * 32 * 16) {
    int e = i & 15, lane = (i >> 4) & 31, ks = (i >> 9) & 1, ntile = i >> 10;
    int K = ks * 32 + ((e >> 3) << 4) + ((lane >> 4) << 3) + (e & 7);
    int N = ntile * 16 + (lane & 15);
    W1p[i] = (K < 38) ? (_Float16)Wt1[K * 256 + N] : (_Float16)0.0f;
  } else {
    int j = i - 16 * 2 * 32 * 16;
    if (j < 3 * 8 * 32 * 16) {
      int e = j & 15, lane = (j >> 4) & 31, ks = (j >> 9) & 7, ntile = j >> 12;
      int K = ks * 32 + ((e >> 3) << 4) + ((lane >> 4) << 3) + (e & 7);
      int N = ntile * 16 + (lane & 15);
      W2p[j] = (_Float16)Wt2[K * 48 + N];
    }
  }
}

// ---------------- big time-varying MLP, fused 2 GEMMs via WMMA ----------------
__global__ __launch_bounds__(256) void mlp_kernel(
    const float* __restrict__ x, const float* __restrict__ xc,
    const _Float16* __restrict__ W1p, const float* __restrict__ bt1,
    const _Float16* __restrict__ W2p, const float* __restrict__ bt2,
    float* __restrict__ fiA, float* __restrict__ fmA, float* __restrict__ EvA,
    int ntiles) {
  __shared__ _Float16 As[16][64];    // input tile, K padded to 64
  __shared__ _Float16 Hs[16][256];   // hidden after tanh

  const int m0   = blockIdx.x * 16;      // first global row of tile
  const int tid  = threadIdx.x;
  const int lane = tid & 31;
  const int wave = tid >> 5;

  // prefetch next tile's dynamic features (speculative; emits global_prefetch_b8)
  if ((int)blockIdx.x + 1 < ntiles && tid < 16)
    __builtin_prefetch(&x[(m0 + 16 + tid) * 6], 0, 1);

  // stage A tile: cols 0..5 from x, 6..37 from xc, rest zero
  for (int i = 0; i < 4; ++i) {
    int e = tid + 256 * i;
    int row = e >> 6, col = e & 63;
    int r = m0 + row;
    float v = 0.0f;
    if (col < 6)       v = x[r * 6 + col];
    else if (col < 38) v = xc[(r % NS) * NGc + (col - 6)];
    As[row][col] = (_Float16)v;
  }
  __syncthreads();

  // ---- GEMM1: [16x64] x [64x256] -> tanh -> Hs ----
  for (int i = 0; i < 2; ++i) {
    int ntile = wave + 8 * i;
    int n0 = ntile * 16;
    v8f acc = {};
    #pragma unroll
    for (int ks = 0; ks < 2; ++ks) {
      v16h a;
      {
        int row  = lane & 15;
        int koff = (lane >> 4) << 3;          // +8 for lanes 16..31
        #pragma unroll
        for (int e = 0; e < 16; ++e) {
          int v2 = e >> 1, hi = e & 1;
          int k = ks * 32 + ((v2 >> 2) << 4) + koff + ((v2 & 3) << 1) + hi;
          a[e] = As[row][k];
        }
      }
      v16h b = *(const v16h*)(W1p + (size_t)(((ntile * 2 + ks) * 32 + lane) * 16));
      acc = __builtin_amdgcn_wmma_f32_16x16x32_f16(false, a, false, b, (short)0, acc, false, false);
    }
    int col   = n0 + (lane & 15);
    int rbase = (lane >> 4) << 3;
    float bias = bt1[col];
    #pragma unroll
    for (int e = 0; e < 8; ++e)
      Hs[rbase + e][col] = (_Float16)tanhf(acc[e] + bias);
  }
  __syncthreads();

  // ---- GEMM2: [16x256] x [256x48] -> activations -> fi/fm/Ev ----
  if (wave < 3) {
    int n0 = wave * 16;
    v8f acc = {};
    #pragma unroll
    for (int ks = 0; ks < 8; ++ks) {
      v16h a;
      {
        int row  = lane & 15;
        int koff = (lane >> 4) << 3;
        #pragma unroll
        for (int e = 0; e < 16; ++e) {
          int v2 = e >> 1, hi = e & 1;
          int k = ks * 32 + ((v2 >> 2) << 4) + koff + ((v2 & 3) << 1) + hi;
          a[e] = Hs[row][k];
        }
      }
      v16h b = *(const v16h*)(W2p + (size_t)(((wave * 8 + ks) * 32 + lane) * 16));
      acc = __builtin_amdgcn_wmma_f32_16x16x32_f16(false, a, false, b, (short)0, acc, false, false);
    }
    int col = n0 + (lane & 15);          // 0..47  = h*3 + c
    int h = col / 3, c = col % 3;
    float bias = bt2[col];
    int rbase = (lane >> 4) << 3;
    #pragma unroll
    for (int e = 0; e < 8; ++e) {
      int r = m0 + rbase + e;
      float val = acc[e] + bias;
      int idx = r * NHb + h;
      if (c == 0)      fiA[idx] = sigmoidf_(val);
      else if (c == 1) fmA[idx] = fmaxf(val, 0.0f);
      else             EvA[idx] = x[r * 6 + 1] * expf(val);   // Evp * exp(v2)
    }
  }
}

// ---------------- static gate MLP: one block per basin ----------------
__global__ __launch_bounds__(256) void gate_kernel(
    const float* __restrict__ xc, const float* __restrict__ Ww1, const float* __restrict__ bw1,
    const float* __restrict__ Ww2, const float* __restrict__ bw2,
    float* kp, float* ksf, float* kg, float* gr, float* gL, float* qb, float* ga) {
  __shared__ float xcs[32];
  __shared__ float Hsm[256];
  __shared__ float exs[16];
  int s = blockIdx.x, tid = threadIdx.x;
  if (tid < 32) xcs[tid] = xc[s * 32 + tid];
  __syncthreads();
  float acc = bw1[tid];
  for (int k = 0; k < 32; ++k) acc += xcs[k] * Ww1[k * 256 + tid];
  Hsm[tid] = tanhf(acc);
  __syncthreads();
  if (tid < 112) {
    float o = bw2[tid];
    for (int j = 0; j < 256; ++j) o += Hsm[j] * Ww2[j * 112 + tid];
    int h = tid / 7, c = tid % 7;
    int idx = s * 16 + h;
    switch (c) {
      case 0: kp[idx]  = sigmoidf_(o); break;
      case 1: ksf[idx] = sigmoidf_(o); break;
      case 2: kg[idx]  = sigmoidf_(o); break;
      case 3: gr[idx]  = sigmoidf_(o); break;
      case 4: gL[idx]  = expf(o);      break;
      case 5: qb[idx]  = fmaxf(o, 0.0f); break;
      case 6: exs[h]   = o;            break;
    }
  }
  __syncthreads();
  if (tid < 16) {
    float mx = -__builtin_inff();
    for (int h = 0; h < 16; ++h) mx = fmaxf(mx, exs[h]);
    float sum = 0.0f;
    for (int h = 0; h < 16; ++h) sum += expf(exs[h] - mx);
    ga[s * 16 + tid] = expf(exs[tid] - mx) / sum;
  }
}

// ---------------- routing MLP + per-bucket softmax over lags ----------------
__global__ __launch_bounds__(256) void route_kernel(
    const float* __restrict__ xc, const float* __restrict__ Wr1, const float* __restrict__ br1,
    const float* __restrict__ Wr2, const float* __restrict__ br2, float* __restrict__ r) {
  __shared__ float xcs[32];
  __shared__ float Hsm[256];
  __shared__ float os[256];
  int s = blockIdx.x, tid = threadIdx.x;
  if (tid < 32) xcs[tid] = xc[s * 32 + tid];
  __syncthreads();
  float acc = br1[tid];
  for (int k = 0; k < 32; ++k) acc += xcs[k] * Wr1[k * 256 + tid];
  Hsm[tid] = tanhf(acc);
  __syncthreads();
  float o = br2[tid];
  for (int j = 0; j < 256; ++j) o += Hsm[j] * Wr2[j * 256 + tid];
  os[tid] = o;
  __syncthreads();
  int g = (tid >> 4) << 4;   // softmax within group of NR=16
  float mx = -__builtin_inff();
  for (int j = 0; j < 16; ++j) mx = fmaxf(mx, os[g + j]);
  float sum = 0.0f;
  for (int j = 0; j < 16; ++j) sum += expf(os[g + j] - mx);
  r[s * 256 + tid] = expf(o - mx) / sum;
}

// ---------------- sequential bucket recurrence: one thread per (s,h) ----------------
__global__ __launch_bounds__(256) void scan_kernel(
    const float* __restrict__ x, const float* __restrict__ fiA, const float* __restrict__ fmA,
    const float* __restrict__ EvA, const float* __restrict__ kp, const float* __restrict__ ksf,
    const float* __restrict__ kg, const float* __restrict__ gr, const float* __restrict__ gL,
    const float* __restrict__ qb, float* __restrict__ Q) {
  int g = blockIdx.x * blockDim.x + threadIdx.x;
  if (g >= NS * NHb) return;
  int s = g >> 4, h = g & 15;
  float kp_ = kp[g], ksf_ = ksf[g], kg_ = kg[g], gr_ = gr[g], gL_ = gL[g], qb_ = qb[g];
  float Sf = 0.0f, Ss = 0.0f, Sd = 0.0f;
  for (int t = 0; t < NT; ++t) {
    int bx = t * NS + s;
    float Prcp = x[bx * 6 + 0], T1 = x[bx * 6 + 2], T2 = x[bx * 6 + 3];
    float vp = sigmoidf_(0.5f * (T1 + T2));
    float ps = Prcp * (1.0f - vp), pl = Prcp * vp;
    int idx = bx * 16 + h;
    float fi_t = fiA[idx], fm_t = fmA[idx], ev_t = EvA[idx];
    float S1 = Sf + ps;
    float qf = fminf(S1, fm_t);
    Sf = S1 - qf;
    float S2 = fmaxf(Ss + pl * fi_t + qf - ev_t, 0.0f);
    float qp = fmaxf(S2 - gL_, 0.0f) * kp_;
    float qs = fminf(S2, gL_) * ksf_;
    Ss = S2 - qp - qs;
    float S3 = Sd + qs * gr_;
    float qd = S3 * kg_ + qb_;
    Sd = fmaxf(S3 - qd, 0.0f);
    Q[idx] = qp + qs * (1.0f - gr_) + qd;
  }
}

// ---------------- 16-tap causal FIR + bucket mixture ----------------
__global__ __launch_bounds__(256) void conv_kernel(
    const float* __restrict__ Q, const float* __restrict__ r, const float* __restrict__ ga,
    float* __restrict__ out) {
  int idx = blockIdx.x * blockDim.x + threadIdx.x;
  const int TO = NT - WARM;
  if (idx >= TO * NS) return;
  int t = WARM + idx / NS;
  int s = idx % NS;
  float acc = 0.0f;
  for (int h = 0; h < 16; ++h) {
    float gah = ga[s * 16 + h];
    float a = 0.0f;
    #pragma unroll
    for (int j = 0; j < 16; ++j)
      a += Q[((t - j) * NS + s) * 16 + h] * r[s * 256 + h * 16 + j];
    acc += a * gah;
  }
  out[idx] = acc;
}

extern "C" void kernel_launch(void* const* d_in, const int* in_sizes, int n_in,
                              void* d_out, int out_size, void* d_ws, size_t ws_size,
                              hipStream_t stream) {
  const float* x   = (const float*)d_in[0];
  const float* xc  = (const float*)d_in[1];
  const float* Wr1 = (const float*)d_in[2];
  const float* br1 = (const float*)d_in[3];
  const float* Wr2 = (const float*)d_in[4];
  const float* br2 = (const float*)d_in[5];
  const float* Ww1 = (const float*)d_in[6];
  const float* bw1 = (const float*)d_in[7];
  const float* Ww2 = (const float*)d_in[8];
  const float* bw2 = (const float*)d_in[9];
  const float* Wt1 = (const float*)d_in[10];
  const float* bt1 = (const float*)d_in[11];
  const float* Wt2 = (const float*)d_in[12];
  const float* bt2 = (const float*)d_in[13];

  char* ws = (char*)d_ws;
  const size_t BIG = (size_t)NT * NS * NHb * sizeof(float);   // 32 MB each
  _Float16* W1p = (_Float16*)(ws);              // 32768 B, fragment-swizzled
  _Float16* W2p = (_Float16*)(ws + 32768);      // 24576 B, fragment-swizzled
  size_t off = 65536;
  float* fiA = (float*)(ws + off); off += BIG;
  float* fmA = (float*)(ws + off); off += BIG;
  float* EvA = (float*)(ws + off); off += BIG;
  float* Q   = (float*)(ws + off); off += BIG;
  float* kp  = (float*)(ws + off); off += NS * NHb * sizeof(float);
  float* ksf = (float*)(ws + off); off += NS * NHb * sizeof(float);
  float* kg  = (float*)(ws + off); off += NS * NHb * sizeof(float);
  float* gr  = (float*)(ws + off); off += NS * NHb * sizeof(float);
  float* gL  = (float*)(ws + off); off += NS * NHb * sizeof(float);
  float* qb  = (float*)(ws + off); off += NS * NHb * sizeof(float);
  float* ga  = (float*)(ws + off); off += NS * NHb * sizeof(float);
  float* rwt = (float*)(ws + off); off += (size_t)NS * NHb * NRl * sizeof(float);

  const int ntiles = (NT * NS) / 16;   // 31250
  prep_weights<<<(16 * 2 * 32 * 16 + 3 * 8 * 32 * 16 + 255) / 256, 256, 0, stream>>>(Wt1, Wt2, W1p, W2p);
  gate_kernel<<<NS, 256, 0, stream>>>(xc, Ww1, bw1, Ww2, bw2, kp, ksf, kg, gr, gL, qb, ga);
  route_kernel<<<NS, 256, 0, stream>>>(xc, Wr1, br1, Wr2, br2, rwt);
  mlp_kernel<<<ntiles, 256, 0, stream>>>(x, xc, W1p, bt1, W2p, bt2, fiA, fmA, EvA, ntiles);
  scan_kernel<<<(NS * NHb + 255) / 256, 256, 0, stream>>>(x, fiA, fmA, EvA, kp, ksf, kg, gr, gL, qb, Q);
  conv_kernel<<<((NT - WARM) * NS + 255) / 256, 256, 0, stream>>>(Q, rwt, ga, (float*)d_out);
}
